// MultiScaleRetention_15702400434237
// MI455X (gfx1250) — matrix operand order repaired and checked by reading, not compiled, not run on hardware
//
#include <hip/hip_runtime.h>
#include <hip/hip_bf16.h>
#include <math.h>

// Problem constants
#define BB 8
#define SS 1024
#define DD 512
#define HH 8
#define DHH 64
#define MM (BB * SS) // 8192 rows

typedef __bf16 bf16_t;
typedef bf16_t v16bf __attribute__((ext_vector_type(16)));
typedef float  v8f   __attribute__((ext_vector_type(8)));
typedef unsigned short u16x8 __attribute__((ext_vector_type(8)));

union FragBF { v16bf v; u16x8 h[2]; };

__device__ __forceinline__ unsigned short f2bf(float f) {
    unsigned int u = __float_as_uint(f);
    u += 0x7FFFu + ((u >> 16) & 1u);   // round-to-nearest-even
    return (unsigned short)(u >> 16);
}

// ---------------------------------------------------------------------------
// Kernel 1: X fp32 -> bf16
// ---------------------------------------------------------------------------
__global__ void convert_x_kernel(const float* __restrict__ X,
                                 unsigned short* __restrict__ Xbf) {
    int i = blockIdx.x * blockDim.x + threadIdx.x;
    if (i < MM * DD) Xbf[i] = f2bf(X[i]);
}

// ---------------------------------------------------------------------------
// Kernel 2: weights -> bf16, transposed to [n][k] row-major.
// which = 0..2: W_{Q,K,V} shape [H,D,DH] -> n = h*64+e, k = d
// which = 3..4: W_{G,O}   shape [D,D]    -> n = col,    k = row
// ---------------------------------------------------------------------------
__global__ void prep_w_kernel(const float* __restrict__ WQ,
                              const float* __restrict__ WK,
                              const float* __restrict__ WV,
                              const float* __restrict__ WG,
                              const float* __restrict__ WO,
                              unsigned short* __restrict__ Wt) {
    int i = blockIdx.x * blockDim.x + threadIdx.x;   // 0 .. 512*512-1
    int which = blockIdx.y;
    int n = i / DD, k = i % DD;
    float val;
    if (which <= 2) {
        const float* W = (which == 0) ? WQ : (which == 1) ? WK : WV;
        int h = n / DHH, e = n % DHH;
        val = W[(h * DD + k) * DHH + e];
    } else {
        const float* W = (which == 3) ? WG : WO;
        val = W[k * DD + n];
    }
    Wt[(size_t)which * DD * DD + (size_t)n * DD + k] = f2bf(val);
}

// ---------------------------------------------------------------------------
// Kernel 3: fused projections.  grid = (M/16, D/256, 4), block = 128 (4 waves)
// Each wave computes a 16x64 strip (4 C fragments) reusing one A fragment.
// which(blockIdx.z): 0=Q(+xpos), 1=K(+xpos downscale), 2=V, 3=G(+swish)
// ---------------------------------------------------------------------------
__global__ __launch_bounds__(128)
void proj_kernel(const unsigned short* __restrict__ Xbf,
                 const unsigned short* __restrict__ Wt,
                 unsigned short* __restrict__ Qbf,
                 unsigned short* __restrict__ Kbf,
                 unsigned short* __restrict__ Vbf,
                 float* __restrict__ SwishG) {
    __shared__ unsigned short As[16 * DD];           // 16 KB A tile
    const int m0 = blockIdx.x * 16;
    const int which = blockIdx.z;
    const int tid = threadIdx.x;

    {   // Stage 16 contiguous rows of Xbf into LDS.
        const u16x8* src = (const u16x8*)(Xbf + (size_t)m0 * DD);
        u16x8* dst = (u16x8*)As;
        for (int i = tid; i < 16 * DD / 8; i += 128) dst[i] = src[i];
    }
    __syncthreads();

    const int wave = tid >> 5, lane = tid & 31;
    const int lrow = lane & 15, lhi = lane >> 4;
    const int ncol = blockIdx.y * 256 + wave * 64;
    const unsigned short* W = Wt + (size_t)which * DD * DD;

    v8f c[4] = {};
    for (int k0 = 0; k0 < DD; k0 += 32) {
        FragBF a;
        // A: 16x32 bf16 — lane holds row lrow; K = lhi*16 + {0..15}
        const unsigned short* ap = As + lrow * DD + k0 + lhi * 16;
        a.h[0] = *(const u16x8*)ap;
        a.h[1] = *(const u16x8*)(ap + 8);
        #pragma unroll
        for (int j = 0; j < 4; ++j) {
            // B: 32x16 bf16 — lane holds col; K = lhi*16 + {0..15}
            FragBF b;
            const unsigned short* bp =
                W + (size_t)(ncol + j * 16 + lrow) * DD + k0 + lhi * 16;
            b.h[0] = *(const u16x8*)bp;
            b.h[1] = *(const u16x8*)(bp + 8);
            c[j] = __builtin_amdgcn_wmma_f32_16x16x32_bf16(false, a.v, false, b.v,
                                                           (short)0, c[j], false, false);
        }
    }

    // Epilogue: C frag VGPR r -> row m0 + r + (lane>=16 ? 8 : 0)
    #pragma unroll
    for (int j = 0; j < 4; ++j) {
        #pragma unroll
        for (int r = 0; r < 8; ++r) {
            int m = m0 + r + (lhi ? 8 : 0);
            int n = ncol + j * 16 + lrow;
            float val = c[j][r];
            if (which <= 1) {
                int s = m & (SS - 1), bidx = m >> 10;
                int h = n >> 6, e = n & 63, jj = e >> 1;
                float sv    = (2.0f * (float)jj + 0.4f * (float)DHH) / (1.4f * (float)DHH);
                float scale = __powf(sv, (float)s * (1.0f / 512.0f));
                if (which == 1) scale = 1.0f / scale;           // xpos downscale for K
                float invf = __powf(10000.0f, -(float)jj * (1.0f / 32.0f));
                float ang  = (float)s * invf;
                float sn = __sinf(ang) * scale, cs = __cosf(ang) * scale;
                // rotate_every_two: pair partner lives in the adjacent lane (n^1)
                float partner = __shfl_xor(val, 1, 32);
                float rot = (e & 1) ? partner : -partner;
                float out = val * cs + rot * sn;
                unsigned short* dst = (which == 0) ? Qbf : Kbf;
                dst[((size_t)(bidx * HH + h) * SS + s) * DHH + e] = f2bf(out);
            } else if (which == 2) {
                int s = m & (SS - 1), bidx = m >> 10;
                int h = n >> 6, e = n & 63;
                Vbf[((size_t)(bidx * HH + h) * SS + s) * DHH + e] = f2bf(val);
            } else {
                float sw = val / (1.0f + __expf(-val));         // swish(g)
                SwishG[(size_t)m * DD + n] = sw;
            }
        }
    }
}

// ---------------------------------------------------------------------------
// Kernel 4: retention attention, flash-style online softmax with post-exp
// decay weighting.  grid = (S/128, B*H), block = 256 (8 waves, 16 q-rows each)
// K tile is staged global->LDS with GLOBAL_LOAD_ASYNC_TO_LDS_B128 (ASYNCcnt).
// ---------------------------------------------------------------------------
__global__ __launch_bounds__(256)
void attn_kernel(const unsigned short* __restrict__ Qbf,
                 const unsigned short* __restrict__ Kbf,
                 const unsigned short* __restrict__ Vbf,
                 float* __restrict__ Y) {
    __shared__ unsigned short Ks[32 * DHH];      // K tile [t][e]     (4 KB)
    __shared__ unsigned short Vs[DHH * 32];      // V tile [e][t]     (4 KB, transposed)
    __shared__ unsigned short Ps[8 * 16 * 32];   // per-wave P' stage (8 KB)

    const int bh = blockIdx.y;
    const int bidx = bh >> 3, h = bh & 7;
    const int tid = threadIdx.x, wave = tid >> 5, lane = tid & 31;
    const int lrow = lane & 15, lhi = lane >> 4;
    const int q0 = blockIdx.x * 128 + wave * 16;
    const size_t head_base = (size_t)bh * SS * DHH;

    // per-head decay: gamma_h = 1 - exp(linspace(log(1/32), log(1/512), 8))
    float gamma = 1.0f - __expf(-logf(32.0f) - (float)h * (logf(16.0f) / 7.0f));
    float lg = __logf(gamma);

    // Q fragments for this wave's 16 rows (k-steps e=0..31 and e=32..63)
    FragBF aq[2];
    {
        const unsigned short* qp = Qbf + head_base + (size_t)(q0 + lrow) * DHH;
        aq[0].h[0] = *(const u16x8*)(qp + lhi * 16);
        aq[0].h[1] = *(const u16x8*)(qp + lhi * 16 + 8);
        aq[1].h[0] = *(const u16x8*)(qp + 32 + lhi * 16);
        aq[1].h[1] = *(const u16x8*)(qp + 32 + lhi * 16 + 8);
    }

    float m_run[8], l_run[8];
    v8f o[4] = {};
    #pragma unroll
    for (int r = 0; r < 8; ++r) { m_run[r] = -1e30f; l_run[r] = 0.0f; }
    unsigned short* Pw = Ps + wave * (16 * 32);
    const unsigned int ks_lds_base = (unsigned int)(unsigned long long)(void*)Ks;

    for (int t0 = 0; t0 < SS; t0 += 32) {
        __syncthreads();   // WAR guard on Ks/Vs
        if (wave == 0) {
            // K tile: 32x64 bf16 = 4 KB, contiguous.  8 wave-wide async B128
            // copies: lane l of copy i moves bytes [ (i*32+l)*16, +16 ).
            const unsigned long long kg =
                (unsigned long long)(Kbf + head_base + (size_t)t0 * DHH);
            #pragma unroll
            for (int i = 0; i < 8; ++i) {
                unsigned int loff = ks_lds_base + (unsigned int)((i * 32 + lane) * 16);
                unsigned long long ga = kg + (unsigned long long)((i * 32 + lane) * 16);
                asm volatile("global_load_async_to_lds_b128 %0, %1, off"
                             :: "v"(loff), "v"(ga) : "memory");
            }
        }
        {   // V: load row-major, store transposed [e][t] (regular VMEM + DS)
            const u16x8* vsrc = (const u16x8*)(Vbf + head_base + (size_t)t0 * DHH);
            u16x8 vv = vsrc[tid];
            int t = (tid * 8) / DHH, e0 = (tid * 8) % DHH;
            #pragma unroll
            for (int i = 0; i < 8; ++i) Vs[(e0 + i) * 32 + t] = vv[i];
        }
        if (wave == 0) asm volatile("s_wait_asynccnt 0x0" ::: "memory");
        __syncthreads();

        if (t0 + 32 < SS) {   // global_prefetch_b8 for next tiles
            __builtin_prefetch(Kbf + head_base + (size_t)(t0 + 32) * DHH, 0, 0);
            __builtin_prefetch(Vbf + head_base + (size_t)(t0 + 32) * DHH, 0, 0);
        }

        // scores: two 16x16 tiles (cols t0..t0+15, t0+16..t0+31), K over e=0..63
        v8f s0 = {}, s1 = {};
        #pragma unroll
        for (int ks = 0; ks < 2; ++ks) {
            FragBF b0, b1;
            const unsigned short* kp0 = Ks + lrow * DHH + ks * 32 + lhi * 16;
            b0.h[0] = *(const u16x8*)kp0;
            b0.h[1] = *(const u16x8*)(kp0 + 8);
            const unsigned short* kp1 = Ks + (16 + lrow) * DHH + ks * 32 + lhi * 16;
            b1.h[0] = *(const u16x8*)kp1;
            b1.h[1] = *(const u16x8*)(kp1 + 8);
            s0 = __builtin_amdgcn_wmma_f32_16x16x32_bf16(false, aq[ks].v, false, b0.v,
                                                         (short)0, s0, false, false);
            s1 = __builtin_amdgcn_wmma_f32_16x16x32_bf16(false, aq[ks].v, false, b1.v,
                                                         (short)0, s1, false, false);
        }

        // online softmax per row + decay-weighted P', staged to LDS as bf16
        #pragma unroll
        for (int r = 0; r < 8; ++r) {
            float v0 = s0[r], v1 = s1[r];
            float mx = fmaxf(v0, v1);
            #pragma unroll
            for (int d = 1; d < 16; d <<= 1) mx = fmaxf(mx, __shfl_xor(mx, d, 16));
            float mn = fmaxf(m_run[r], mx);
            float alpha = __expf(m_run[r] - mn);
            float p0 = __expf(v0 - mn), p1 = __expf(v1 - mn);
            float rs = p0 + p1;
            #pragma unroll
            for (int d = 1; d < 16; d <<= 1) rs += __shfl_xor(rs, d, 16);
            l_run[r] = l_run[r] * alpha + rs;
            m_run[r] = mn;
            #pragma unroll
            for (int j = 0; j < 4; ++j) o[j][r] = o[j][r] * alpha;
            int qrow = q0 + r + (lhi ? 8 : 0);
            float w0 = __expf(lg * fabsf((float)(qrow - (t0 + lrow))));
            float w1 = __expf(lg * fabsf((float)(qrow - (t0 + 16 + lrow))));
            int prow = r + (lhi ? 8 : 0);
            Pw[prow * 32 + lrow]      = f2bf(p0 * w0);
            Pw[prow * 32 + 16 + lrow] = f2bf(p1 * w1);
        }

        // O += P'(16x32) @ V(32x64)  — 4 WMMAs
        FragBF ap;
        const unsigned short* pp = Pw + lrow * 32 + lhi * 16;
        ap.h[0] = *(const u16x8*)pp;
        ap.h[1] = *(const u16x8*)(pp + 8);
        #pragma unroll
        for (int j = 0; j < 4; ++j) {
            FragBF bv;
            const unsigned short* vp = Vs + (j * 16 + lrow) * 32 + lhi * 16;
            bv.h[0] = *(const u16x8*)vp;
            bv.h[1] = *(const u16x8*)(vp + 8);
            o[j] = __builtin_amdgcn_wmma_f32_16x16x32_bf16(false, ap.v, false, bv.v,
                                                           (short)0, o[j], false, false);
        }
    }

    // Normalize and write Y as [B, S, D] fp32 (col = h*64 + e).
    #pragma unroll
    for (int r = 0; r < 8; ++r) {
        int srow = q0 + r + (lhi ? 8 : 0);
        float inv_l = 1.0f / l_run[r];
        #pragma unroll
        for (int j = 0; j < 4; ++j) {
            int col = h * DHH + j * 16 + lrow;
            Y[((size_t)bidx * SS + srow) * DD + col] = o[j][r] * inv_l;
        }
    }
}

// ---------------------------------------------------------------------------
// Kernel 5: GroupNorm(H groups) + swish gating -> Z bf16.  grid = M, block = 512
// ---------------------------------------------------------------------------
__global__ __launch_bounds__(512)
void gn_gate_kernel(const float* __restrict__ Y,
                    const float* __restrict__ SwishG,
                    const float* __restrict__ gnw,
                    const float* __restrict__ gnb,
                    unsigned short* __restrict__ Zbf) {
    __shared__ float rs[16], rq[16];
    const int row = blockIdx.x;
    const int c = threadIdx.x;
    const int wave = c >> 5, lane = c & 31;
    float y = Y[(size_t)row * DD + c];
    float s = y, q = y * y;
    #pragma unroll
    for (int d = 1; d < 32; d <<= 1) {
        s += __shfl_xor(s, d, 32);
        q += __shfl_xor(q, d, 32);
    }
    if (lane == 0) { rs[wave] = s; rq[wave] = q; }
    __syncthreads();
    int g = c >> 6;                               // group = head (64 cols = 2 waves)
    float sum = rs[2 * g] + rs[2 * g + 1];
    float sq  = rq[2 * g] + rq[2 * g + 1];
    float mu  = sum * (1.0f / 64.0f);
    float var = sq * (1.0f / 64.0f) - mu * mu;
    float yn  = (y - mu) * rsqrtf(var + 1e-5f);
    yn = yn * gnw[c] + gnb[c];
    float z = SwishG[(size_t)row * DD + c] * yn;
    Zbf[(size_t)row * DD + c] = f2bf(z);
}

// ---------------------------------------------------------------------------
// Kernel 6: Out = Z @ W_O  (fp32 out).  grid = (M/16, D/256), block = 128.
// Same 16x64-per-wave register blocking as proj_kernel.
// ---------------------------------------------------------------------------
__global__ __launch_bounds__(128)
void out_gemm_kernel(const unsigned short* __restrict__ Zbf,
                     const unsigned short* __restrict__ WOt,
                     float* __restrict__ Out) {
    __shared__ unsigned short As[16 * DD];
    const int m0 = blockIdx.x * 16;
    const int tid = threadIdx.x;
    {
        const u16x8* src = (const u16x8*)(Zbf + (size_t)m0 * DD);
        u16x8* dst = (u16x8*)As;
        for (int i = tid; i < 16 * DD / 8; i += 128) dst[i] = src[i];
    }
    __syncthreads();
    const int wave = tid >> 5, lane = tid & 31;
    const int lrow = lane & 15, lhi = lane >> 4;
    const int ncol = blockIdx.y * 256 + wave * 64;
    v8f c[4] = {};
    for (int k0 = 0; k0 < DD; k0 += 32) {
        FragBF a;
        const unsigned short* ap = As + lrow * DD + k0 + lhi * 16;
        a.h[0] = *(const u16x8*)ap;
        a.h[1] = *(const u16x8*)(ap + 8);
        #pragma unroll
        for (int j = 0; j < 4; ++j) {
            FragBF b;
            const unsigned short* bp =
                WOt + (size_t)(ncol + j * 16 + lrow) * DD + k0 + lhi * 16;
            b.h[0] = *(const u16x8*)bp;
            b.h[1] = *(const u16x8*)(bp + 8);
            c[j] = __builtin_amdgcn_wmma_f32_16x16x32_bf16(false, a.v, false, b.v,
                                                           (short)0, c[j], false, false);
        }
    }
    #pragma unroll
    for (int j = 0; j < 4; ++j) {
        #pragma unroll
        for (int r = 0; r < 8; ++r) {
            int m = m0 + r + (lhi ? 8 : 0);
            Out[(size_t)m * DD + ncol + j * 16 + lrow] = c[j][r];
        }
    }
}

// ---------------------------------------------------------------------------
extern "C" void kernel_launch(void* const* d_in, const int* in_sizes, int n_in,
                              void* d_out, int out_size, void* d_ws, size_t ws_size,
                              hipStream_t stream) {
    const float* X   = (const float*)d_in[0];
    const float* WQ  = (const float*)d_in[1];
    const float* WK  = (const float*)d_in[2];
    const float* WV  = (const float*)d_in[3];
    const float* WG  = (const float*)d_in[4];
    const float* WO  = (const float*)d_in[5];
    const float* gnw = (const float*)d_in[6];
    const float* gnb = (const float*)d_in[7];

    char* ws = (char*)d_ws;
    const size_t MD2 = (size_t)MM * DD * 2;   // 8 MB  (bf16 matrix)
    const size_t W2  = (size_t)DD * DD * 2;   // 512 KB (bf16 weight)
    const size_t MD4 = (size_t)MM * DD * 4;   // 16 MB (fp32 matrix)

    size_t off = 0;
    unsigned short* Xbf = (unsigned short*)(ws + off); off += MD2;
    unsigned short* Wt  = (unsigned short*)(ws + off); off += 5 * W2; // Q,K,V,G,O
    unsigned short* Qbf = (unsigned short*)(ws + off); off += MD2;
    unsigned short* Kbf = (unsigned short*)(ws + off); off += MD2;
    unsigned short* Vbf = (unsigned short*)(ws + off); off += MD2;
    float*          Gsw = (float*)(ws + off);          off += MD4;
    float*          Yf  = (float*)(ws + off);          off += MD4;
    unsigned short* Zbf = (unsigned short*)(ws + off); off += MD2;
    unsigned short* WOt = Wt + 4 * (size_t)DD * DD;
    float* Out = (float*)d_out;

    convert_x_kernel<<<dim3(MM * DD / 256), dim3(256), 0, stream>>>(X, Xbf);
    prep_w_kernel<<<dim3(DD * DD / 256, 5), dim3(256), 0, stream>>>(WQ, WK, WV, WG, WO, Wt);
    proj_kernel<<<dim3(MM / 16, DD / 256, 4), dim3(128), 0, stream>>>(
        Xbf, Wt, Qbf, Kbf, Vbf, Gsw);
    attn_kernel<<<dim3(SS / 128, BB * HH), dim3(256), 0, stream>>>(Qbf, Kbf, Vbf, Yf);
    gn_gate_kernel<<<dim3(MM), dim3(512), 0, stream>>>(Yf, Gsw, gnw, gnb, Zbf);
    out_gemm_kernel<<<dim3(MM / 16, DD / 256), dim3(128), 0, stream>>>(Zbf, WOt, Out);
}
